// DisparityUpsampler_38250978738251
// MI455X (gfx1250) — matrix operand order repaired and checked by reading, not compiled, not run on hardware
//
#include <hip/hip_runtime.h>
#include <hip/hip_bf16.h>

#define B_ 2
#define H_ 112
#define W_ 256

typedef __attribute__((ext_vector_type(16))) _Float16 v16h;
typedef __attribute__((ext_vector_type(8)))  _Float16 v8h;
typedef __attribute__((ext_vector_type(8)))  float    v8f;
typedef __attribute__((ext_vector_type(4)))  float    v4f;

__device__ inline v8f zero_v8f() {
  v8f r;
#pragma unroll
  for (int i = 0; i < 8; ++i) r[i] = 0.0f;
  return r;
}

// A-fragment: lane holds contiguous K run [K0..K0+15] (matches B's
// contiguous order) -> two contiguous 16B loads merged into one v16h.
__device__ inline v16h load_a16(const _Float16* p) {
  v8h lo = *(const v8h*)p;
  v8h hi = *(const v8h*)(p + 8);
  return __builtin_shufflevector(lo, hi, 0, 1, 2, 3, 4, 5, 6, 7, 8, 9, 10, 11,
                                 12, 13, 14, 15);
}

// ---------------------------------------------------------------------------
// Weight re-layout: f32 OIHW (M, CIN_SRC, 3, 3) -> f16 [tap][M][C] (C fastest)
// ---------------------------------------------------------------------------
template <int COUT, int CIN, int CIN_SRC, int COFF, int TAPS>
__global__ __launch_bounds__(256) void prep_w(const float* __restrict__ src,
                                              _Float16* __restrict__ dst) {
  int i = blockIdx.x * blockDim.x + threadIdx.x;
  if (i >= COUT * CIN * TAPS) return;
  int c = i % CIN;
  int m = (i / CIN) % COUT;
  int tap = i / (CIN * COUT);
  dst[((size_t)tap * COUT + m) * CIN + c] =
      (_Float16)src[((size_t)m * CIN_SRC + (c + COFF)) * TAPS + tap];
}

// BN2 scale/shift, disp-channel weights of c1, and the B-side zero page.
__global__ __launch_bounds__(256) void prep_misc(
    const float* __restrict__ g2, const float* __restrict__ b2,
    const float* __restrict__ m2, const float* __restrict__ v2,
    const float* __restrict__ wc1, float* __restrict__ s2,
    float* __restrict__ t2, float* __restrict__ wd,
    float* __restrict__ zb) {
  int t = blockIdx.x * blockDim.x + threadIdx.x;
  if (t < 256) {
    float s = g2[t] * rsqrtf(v2[t] + 1e-5f);
    s2[t] = s;
    t2[t] = b2[t] - m2[t] * s;
    zb[t] = 0.f;  // 1KB zero page for out-of-bounds B columns
  }
  if (t < 2304) {
    int m = t / 9, tap = t % 9;
    wd[t] = wc1[((size_t)m * 257) * 9 + tap];  // input channel 0 of concat
  }
}

// ---------------------------------------------------------------------------
// enc1: conv 3->256 k4 s4 (K=48, 0.5% of FLOPs) + BN + ReLU -> f16 NHWC.
// ---------------------------------------------------------------------------
__global__ __launch_bounds__(256) void enc1_kernel(
    const float* __restrict__ patch, const float* __restrict__ w,
    const float* __restrict__ g, const float* __restrict__ bb,
    const float* __restrict__ mm, const float* __restrict__ vv,
    _Float16* __restrict__ feat1) {
  __shared__ float sh[48];
  int n = blockIdx.x;  // 0 .. B_*H_*W_-1
  int x = n % W_;
  int r = n / W_;
  int y = r % H_;
  int b = r / H_;
  int t = threadIdx.x;
  if (t < 48) {
    int ci = t >> 4;
    int ky = (t >> 2) & 3;
    int kx = t & 3;
    sh[t] = patch[(((size_t)b * 3 + ci) * (4 * H_) + 4 * y + ky) * (4 * W_) +
                  4 * x + kx];
  }
  __syncthreads();
  int m = t;
  const v4f* w4 = (const v4f*)(w + m * 48);  // 16B aligned (m*192 bytes)
  float acc = 0.f;
#pragma unroll
  for (int i = 0; i < 12; ++i) {
    v4f wv = w4[i];
#pragma unroll
    for (int j = 0; j < 4; ++j) acc += sh[i * 4 + j] * wv[j];
  }
  float s = g[m] * rsqrtf(vv[m] + 1e-5f);
  float o = (acc - mm[m]) * s + bb[m];
  o = o > 0.f ? o : 0.f;
  feat1[(size_t)n * 256 + m] = (_Float16)o;
}

// ---------------------------------------------------------------------------
// Implicit-im2col WMMA conv-GEMM.
//   A = Wt  (f16 [TAPS][COUT][CIN]), B = Xin (f16 NHWC)
//   MODE 0: y = acc*P0 + P1, ReLU      -> f16 NHWC   (enc2, BN folded)
//   MODE 1: y = acc + P0 + disp term (P1 = wd[m][9]), ReLU -> f16 (c1)
//   MODE 2: y = acc + P0, ReLU         -> f16 NHWC   (c2)
//   MODE 3: y = acc + P0               -> f32 [n][COUT] (c3, 1x1)
// Per wave: (MT*16) x (NT*16) output tile; B double-buffered over k-steps;
// each A fragment feeds NT WMMAs. 8 waves / block.
// ---------------------------------------------------------------------------
template <int CIN, int COUT, int TAPS, int MT, int NT, int MODE>
__global__ __launch_bounds__(256) void convgemm_k(
    const _Float16* __restrict__ Xin, const _Float16* __restrict__ Wt,
    const float* __restrict__ P0, const float* __restrict__ P1,
    const float* __restrict__ dsp, const _Float16* __restrict__ zbuf,
    _Float16* __restrict__ outh, float* __restrict__ outf) {
  static_assert(CIN % 32 == 0, "");
  static_assert(W_ % (16 * NT) == 0, "");
  constexpr int KC = CIN / 32;
  constexpr int MG = COUT / (16 * MT);
  constexpr int XG = W_ / (16 * NT);
  const int lane = threadIdx.x & 31;
  const int gw = blockIdx.x * (blockDim.x >> 5) + (threadIdx.x >> 5);
  const int total = B_ * H_ * XG * MG;
  if (gw >= total) return;
  const int mg = gw % MG;
  int sg = gw / MG;
  const int xg = sg % XG;
  sg /= XG;
  const int y = sg % H_;
  const int b = sg / H_;
  const int x0 = xg * (16 * NT);
  const int m0 = mg * (16 * MT);
  const int col = lane & 15;
  const int hi4 = lane >> 4;

  v8f acc[MT][NT];
#pragma unroll
  for (int mt = 0; mt < MT; ++mt)
#pragma unroll
    for (int nt = 0; nt < NT; ++nt) acc[mt][nt] = zero_v8f();

  for (int tap = 0; tap < TAPS; ++tap) {
    const int dy = (TAPS == 9) ? (tap / 3 - 1) : 0;
    const int dx = (TAPS == 9) ? (tap % 3 - 1) : 0;
    const int yy = y + dy;
    const bool rowok = (unsigned)yy < (unsigned)H_;

    // B pointers: OOB columns read the zero page -> unconditional loads.
    const _Float16* bptr[NT];
#pragma unroll
    for (int nt = 0; nt < NT; ++nt) {
      int xx = x0 + nt * 16 + col + dx;
      bool ok = rowok && (unsigned)xx < (unsigned)W_;
      const _Float16* bp =
          Xin + ((size_t)((b * H_ + yy) * W_ + xx)) * CIN + 16 * hi4;
      bptr[nt] = ok ? bp : (zbuf + 16 * hi4);
    }
    const _Float16* aptr[MT];
#pragma unroll
    for (int mt = 0; mt < MT; ++mt)
      aptr[mt] =
          Wt + ((size_t)tap * COUT + (m0 + 16 * mt + col)) * CIN + 16 * hi4;

    // B fragments double-buffered: next step's B loads issue before the
    // MT*NT WMMAs of the current step.
    v16h bf[NT];
#pragma unroll
    for (int nt = 0; nt < NT; ++nt) bf[nt] = *(const v16h*)(bptr[nt]);

#pragma unroll
    for (int kc = 0; kc < KC; ++kc) {
      v16h bfn[NT];
      if (kc + 1 < KC) {
        const int c1 = (kc + 1) * 32;
#pragma unroll
        for (int nt = 0; nt < NT; ++nt)
          bfn[nt] = *(const v16h*)(bptr[nt] + c1);
      }
#pragma unroll
      for (int mt = 0; mt < MT; ++mt) {
        v16h af = load_a16(aptr[mt] + kc * 32);
#pragma unroll
        for (int nt = 0; nt < NT; ++nt)
          acc[mt][nt] = __builtin_amdgcn_wmma_f32_16x16x32_f16(
              false, af, false, bf[nt], (short)0, acc[mt][nt], false, false);
      }
      if (kc + 1 < KC) {
#pragma unroll
        for (int nt = 0; nt < NT; ++nt) bf[nt] = bfn[nt];
      }
    }
  }

  // disp-channel (concat channel 0) contribution for c1: rank-9 VALU add.
  float dn[NT][9];
  if (MODE == 1) {
#pragma unroll
    for (int nt = 0; nt < NT; ++nt)
#pragma unroll
      for (int tap = 0; tap < 9; ++tap) {
        int yy = y + tap / 3 - 1;
        int xx = x0 + nt * 16 + col + tap % 3 - 1;
        float d = 0.f;
        if ((unsigned)yy < (unsigned)H_ && (unsigned)xx < (unsigned)W_)
          d = dsp[((size_t)b * H_ + yy) * W_ + xx];
        dn[nt][tap] = d;
      }
  }

  // Epilogue: C/D layout -> row m = mb + r, col n = lane%16; 8 consecutive
  // channels per lane -> one 16B (f16) / 32B (f32) store.
#pragma unroll
  for (int mt = 0; mt < MT; ++mt) {
    const int mb = m0 + 16 * mt + 8 * hi4;
    float p0v[8], p1v[8];
#pragma unroll
    for (int i = 0; i < 8; ++i) p0v[i] = P0[mb + i];
    if (MODE == 0) {
#pragma unroll
      for (int i = 0; i < 8; ++i) p1v[i] = P1[mb + i];
    }
#pragma unroll
    for (int nt = 0; nt < NT; ++nt) {
      v8f r = acc[mt][nt];
#pragma unroll
      for (int i = 0; i < 8; ++i) {
        float v = r[i];
        if (MODE == 0)
          v = v * p0v[i] + p1v[i];
        else
          v = v + p0v[i];
        if (MODE == 1) {
#pragma unroll
          for (int tap = 0; tap < 9; ++tap)
            v += P1[(size_t)(mb + i) * 9 + tap] * dn[nt][tap];
        }
        if (MODE != 3) v = v > 0.f ? v : 0.f;
        r[i] = v;
      }
      const size_t n = ((size_t)b * H_ + y) * W_ + x0 + nt * 16 + col;
      if (MODE == 3) {
        *(v8f*)(outf + n * COUT + mb) = r;
      } else {
        v8h h;
#pragma unroll
        for (int i = 0; i < 8; ++i) h[i] = (_Float16)r[i];
        *(v8h*)(outh + n * COUT + mb) = h;
      }
    }
  }
}

// ---------------------------------------------------------------------------
// softmax over q (9) per (pixel, scale-slot) + convex-combination upsample.
// raw layout [n][144], ch = q*16 + (sy*4+sx).
// ---------------------------------------------------------------------------
__global__ __launch_bounds__(256) void softmax_up(
    const float* __restrict__ raw, const float* __restrict__ dsp,
    float* __restrict__ out) {
  int t = blockIdx.x * blockDim.x + threadIdx.x;
  if (t >= B_ * H_ * W_ * 16) return;
  int s = t & 15;
  int n = t >> 4;
  int x = n % W_;
  int r = n / W_;
  int y = r % H_;
  int b = r / H_;
  float v[9];
  float mx = -3.4e38f;
#pragma unroll
  for (int q = 0; q < 9; ++q) {
    v[q] = raw[(size_t)n * 144 + q * 16 + s];
    mx = v[q] > mx ? v[q] : mx;
  }
  float sum = 0.f;
#pragma unroll
  for (int q = 0; q < 9; ++q) {
    v[q] = expf(v[q] - mx);
    sum += v[q];
  }
  float inv = 1.f / sum;
  float up = 0.f;
#pragma unroll
  for (int q = 0; q < 9; ++q) {
    int yy = y + q / 3 - 1;
    int xx = x + q % 3 - 1;
    float d = 0.f;
    if ((unsigned)yy < (unsigned)H_ && (unsigned)xx < (unsigned)W_)
      d = dsp[((size_t)b * H_ + yy) * W_ + xx];
    up += v[q] * inv * d;
  }
  int sy = s >> 2, sx = s & 3;
  out[((size_t)b * (4 * H_) + (4 * y + sy)) * (4 * W_) + 4 * x + sx] = up;
}

// ---------------------------------------------------------------------------
extern "C" void kernel_launch(void* const* d_in, const int* in_sizes, int n_in,
                              void* d_out, int out_size, void* d_ws,
                              size_t ws_size, hipStream_t stream) {
  const float* patch = (const float*)d_in[0];
  const float* disp = (const float*)d_in[1];
  const float* w_enc1 = (const float*)d_in[2];
  const float* bn1_g = (const float*)d_in[3];
  const float* bn1_b = (const float*)d_in[4];
  const float* bn1_m = (const float*)d_in[5];
  const float* bn1_v = (const float*)d_in[6];
  const float* w_enc2 = (const float*)d_in[7];
  const float* bn2_g = (const float*)d_in[8];
  const float* bn2_b = (const float*)d_in[9];
  const float* bn2_m = (const float*)d_in[10];
  const float* bn2_v = (const float*)d_in[11];
  const float* w_c1 = (const float*)d_in[12];
  const float* b_c1 = (const float*)d_in[13];
  const float* w_c2 = (const float*)d_in[14];
  const float* b_c2 = (const float*)d_in[15];
  const float* w_c3 = (const float*)d_in[16];
  const float* b_c3 = (const float*)d_in[17];
  float* out = (float*)d_out;

  char* ws = (char*)d_ws;
  size_t off = 0;
  auto alloc = [&](size_t bytes) -> char* {
    char* p = ws + off;
    off = (off + bytes + 255) & ~(size_t)255;
    return p;
  };
  const size_t NPIX = (size_t)B_ * H_ * W_;  // 57344
  _Float16* feat1 = (_Float16*)alloc(NPIX * 256 * 2);
  _Float16* feat2 = (_Float16*)alloc(NPIX * 256 * 2);
  _Float16* x2 = (_Float16*)alloc(NPIX * 512 * 2);
  float* raw = (float*)alloc(NPIX * 144 * 4);
  _Float16* Wt2 = (_Float16*)alloc((size_t)9 * 256 * 256 * 2);
  _Float16* Wtc1 = (_Float16*)alloc((size_t)9 * 256 * 256 * 2);
  _Float16* Wtc2 = (_Float16*)alloc((size_t)9 * 512 * 256 * 2);
  _Float16* Wtc3 = (_Float16*)alloc((size_t)144 * 512 * 2);
  float* s2 = (float*)alloc(256 * 4);
  float* t2 = (float*)alloc(256 * 4);
  float* wd = (float*)alloc(2304 * 4);
  float* zb = (float*)alloc(256 * 4);  // 1KB zero page (B OOB columns)
  const _Float16* zbuf = (const _Float16*)zb;
  _Float16* x1 = feat1;  // feat1 dead after enc2 -> reuse for c1 output

  // weight re-layouts + folded BN params + zero page
  prep_w<256, 256, 256, 0, 9><<<2304, 256, 0, stream>>>(w_enc2, Wt2);
  prep_w<256, 256, 257, 1, 9><<<2304, 256, 0, stream>>>(w_c1, Wtc1);
  prep_w<512, 256, 256, 0, 9><<<4608, 256, 0, stream>>>(w_c2, Wtc2);
  prep_w<144, 512, 512, 0, 1><<<288, 256, 0, stream>>>(w_c3, Wtc3);
  prep_misc<<<9, 256, 0, stream>>>(bn2_g, bn2_b, bn2_m, bn2_v, w_c1, s2, t2,
                                   wd, zb);
  // encoder stage 1 (VALU)
  enc1_kernel<<<(int)NPIX, 256, 0, stream>>>(patch, w_enc1, bn1_g, bn1_b,
                                             bn1_m, bn1_v, feat1);
  // WMMA conv-GEMM chain: 64x64 tiles (MT=4,NT=4), c3 uses 48x64 (MT=3)
  convgemm_k<256, 256, 9, 4, 4, 0><<<448, 256, 0, stream>>>(
      feat1, Wt2, s2, t2, nullptr, zbuf, feat2, nullptr);
  convgemm_k<256, 256, 9, 4, 4, 1><<<448, 256, 0, stream>>>(
      feat2, Wtc1, b_c1, wd, disp, zbuf, x1, nullptr);
  convgemm_k<256, 512, 9, 4, 4, 2><<<896, 256, 0, stream>>>(
      x1, Wtc2, b_c2, nullptr, nullptr, zbuf, x2, nullptr);
  convgemm_k<512, 144, 1, 3, 4, 3><<<336, 256, 0, stream>>>(
      x2, Wtc3, b_c3, nullptr, nullptr, zbuf, nullptr, raw);
  // softmax + learned 4x upsample
  softmax_up<<<3584, 256, 0, stream>>>(raw, disp, out);
}
